// SetAbstraction_11991548690859
// MI455X (gfx1250) — matrix operand order repaired and checked by reading, not compiled, not run on hardware
//
#include <hip/hip_runtime.h>
#include <math.h>

// Problem constants from the reference
#define Bc 8
#define Nc 4096
#define Sc 1024
#define Kc 128
#define R2c 0.04f
#define NFRAG 28            // 4 (L1) + 8 (L2) + 16 (L3) pre-swizzled B fragments

typedef __attribute__((ext_vector_type(16))) _Float16 v16h;
typedef __attribute__((ext_vector_type(8)))  _Float16 v8h;
typedef __attribute__((ext_vector_type(8)))  float    v8f;

// ---------------------------------------------------------------------------
// Kernel 0: pre-swizzle f32 weights into f16 WMMA B fragments (lane-major).
// Fragment f, lane l, elem i  ->  wfrag[f*512 + l*16 + i]
//   B layout (32x16 f16): n = n0 + (l&15), k = k0 + 16*(l>>4) + i
// ---------------------------------------------------------------------------
__global__ __launch_bounds__(256) void prep_kernel(
    const float* __restrict__ W1, const float* __restrict__ W2,
    const float* __restrict__ W3, _Float16* __restrict__ wfrag)
{
    int e = blockIdx.x * 256 + threadIdx.x;
    if (e >= NFRAG * 512) return;
    int f    = e >> 9;
    int lane = (e >> 4) & 31;
    int i    = e & 15;

    const float* W; int ncols, nt, k0, kmax;
    if (f < 4)       { W = W1; ncols = 64;  nt = f;           k0 = 0;              kmax = 6;  }
    else if (f < 12) { int q = f - 4;  W = W2; ncols = 64;  nt = q >> 1; k0 = (q & 1) * 32; kmax = 64; }
    else             { int q = f - 12; W = W3; ncols = 128; nt = q >> 1; k0 = (q & 1) * 32; kmax = 64; }

    int n = nt * 16 + (lane & 15);
    int k = k0 + ((lane >> 4) << 4) + i;
    wfrag[e] = (_Float16)((k < kmax) ? W[k * ncols + n] : 0.0f);
}

// ---------------------------------------------------------------------------
// Kernel 1: farthest point sampling, one workgroup per cloud.
// ---------------------------------------------------------------------------
__global__ __launch_bounds__(256) void fps_kernel(const float* __restrict__ pos,
                                                  int* __restrict__ fps_idx)
{
    const int b   = blockIdx.x;
    const int tid = threadIdx.x;
    const float* p = pos + (size_t)b * Nc * 3;

    __shared__ float mind[Nc];
    __shared__ float rv[8];
    __shared__ int   ri[8];
    __shared__ int   bcast;

    const float q0 = p[0], q1 = p[1], q2 = p[2];
    for (int i = tid; i < Nc; i += 256) {
        float dx = p[3*i+0]-q0, dy = p[3*i+1]-q1, dz = p[3*i+2]-q2;
        mind[i] = dx*dx + dy*dy + dz*dz;
    }
    if (tid == 0) fps_idx[b*Sc] = 0;
    __syncthreads();

    for (int s = 1; s < Sc; ++s) {
        float best = -1.0f; int bi = Nc;
        for (int i = tid; i < Nc; i += 256) {
            float v = mind[i];
            if (v > best) { best = v; bi = i; }   // first occurrence kept
        }
        for (int o = 16; o > 0; o >>= 1) {        // wave32 argmax, index tiebreak
            float ov = __shfl_down(best, o);
            int   oi = __shfl_down(bi,   o);
            if (ov > best || (ov == best && oi < bi)) { best = ov; bi = oi; }
        }
        if ((tid & 31) == 0) { rv[tid >> 5] = best; ri[tid >> 5] = bi; }
        __syncthreads();
        if (tid == 0) {
            float bv = rv[0]; int bx = ri[0];
            for (int w = 1; w < 8; ++w)
                if (rv[w] > bv || (rv[w] == bv && ri[w] < bx)) { bv = rv[w]; bx = ri[w]; }
            bcast = bx;
            fps_idx[b*Sc + s] = bx;
        }
        __syncthreads();
        const int nxt = bcast;
        const float n0 = p[3*nxt+0], n1 = p[3*nxt+1], n2 = p[3*nxt+2];
        for (int i = tid; i < Nc; i += 256) {
            float dx = p[3*i+0]-n0, dy = p[3*i+1]-n1, dz = p[3*i+2]-n2;
            mind[i] = fminf(mind[i], dx*dx + dy*dy + dz*dz);
        }
        __syncthreads();
    }
}

// ---------------------------------------------------------------------------
// Kernel 2: ball query — exact K smallest masked d^2 (ties -> lowest index).
// 8-bit-digit radix select (4 histogram passes) on float bit patterns in LDS.
// One workgroup per (b,s); out-of-radius fill slots get -1.
// ---------------------------------------------------------------------------
__global__ __launch_bounds__(256) void ball_kernel(const float* __restrict__ pos,
                                                   const int* __restrict__ fps_idx,
                                                   int* __restrict__ nbr)
{
    const int bs  = blockIdx.x;
    const int b   = bs >> 10;           // Sc == 1024
    const int tid = threadIdx.x;
    const float* p = pos + (size_t)b * Nc * 3;

    __shared__ unsigned keys[Nc];
    __shared__ unsigned ttot[256];
    __shared__ unsigned tscan[257];
    __shared__ int      hist[256];
    __shared__ unsigned sPref;
    __shared__ int      sR;
    __shared__ float    sq[3];

    if (tid == 0) {
        int qi = fps_idx[bs];
        sq[0] = p[3*qi+0]; sq[1] = p[3*qi+1]; sq[2] = p[3*qi+2];
        sPref = 0u; sR = Kc;
    }
    __syncthreads();
    const float q0 = sq[0], q1 = sq[1], q2 = sq[2];

    __builtin_prefetch(p, 0, 0);        // global_prefetch_b8 (L2-resident cloud)
    for (int i = tid; i < Nc; i += 256) {
        float dx = p[3*i+0]-q0, dy = p[3*i+1]-q1, dz = p[3*i+2]-q2;
        float d2 = dx*dx + dy*dy + dz*dz;
        keys[i] = (d2 <= R2c) ? __float_as_uint(d2) : 0xFFFFFFFFu;
    }
    __syncthreads();

    // radix select: value T of the Kc-th smallest key, one byte per pass
    for (int shift = 24; shift >= 0; shift -= 8) {
        hist[tid] = 0;
        __syncthreads();
        const unsigned pref   = sPref;
        const unsigned hiMask = (shift == 24) ? 0u : (0xFFFFFFFFu << (shift + 8));
        for (int i = tid; i < Nc; i += 256) {
            unsigned k = keys[i];
            if (((k ^ pref) & hiMask) == 0u)
                atomicAdd(&hist[(k >> shift) & 255u], 1);
        }
        __syncthreads();
        if (tid == 0) {
            int racc = sR;
            unsigned d = 0;
            for (; d < 256; ++d) { int c = hist[d]; if (racc <= c) break; racc -= c; }
            sR = racc;
            sPref = pref | (d << shift);
        }
        __syncthreads();
    }
    const unsigned T = sPref;

    // deterministic compaction: all keys < T, then lowest-index keys == T
    const int base = tid * 16;          // contiguous chunk per thread
    int lc = 0, ec = 0;
    for (int i = 0; i < 16; ++i) {
        unsigned k = keys[base + i];
        lc += (k <  T);
        ec += (k == T);
    }
    ttot[tid] = ((unsigned)lc << 16) | (unsigned)ec;
    __syncthreads();
    if (tid == 0) {
        unsigned acc = 0;
        for (int i = 0; i < 256; ++i) { tscan[i] = acc; acc += ttot[i]; }
        tscan[256] = acc;
    }
    __syncthreads();
    const int totLess = (int)(tscan[256] >> 16);
    const int eqTake  = Kc - totLess;
    int exLess = (int)(tscan[tid] >> 16);
    int exEq   = (int)(tscan[tid] & 0xFFFFu);
    int* dst = nbr + (size_t)bs * Kc;
    for (int i = 0; i < 16; ++i) {
        int gi = base + i;
        unsigned k = keys[gi];
        if (k < T) {
            dst[exLess++] = gi;
        } else if (k == T) {
            if (exEq < eqTake) dst[totLess + exEq] = (k == 0xFFFFFFFFu) ? -1 : gi;
            exEq++;
        }
    }
}

// ---------------------------------------------------------------------------
// WMMA fragment helpers
// ---------------------------------------------------------------------------
__device__ inline v16h zero_v16h() {
    v16h v;
#pragma unroll
    for (int i = 0; i < 16; ++i) v[i] = (_Float16)0.0f;
    return v;
}

// pre-swizzled B fragment: one contiguous 32B LDS read per lane
__device__ inline v16h wget(const _Float16* __restrict__ wlds, int f, int lane)
{
    return *(const v16h*)(wlds + (f << 9) + (lane << 4));
}

// A fragment from a 16x64 row-major f16 LDS tile, k chunk [k0,k0+32):
// two contiguous 16B LDS reads per lane
__device__ inline v16h load_afrag(const _Float16* __restrict__ hb, int k0, int lane)
{
    const int m  = lane & 15;
    const int kb = k0 + ((lane >> 4) << 4);
    const _Float16* p0 = hb + m * 64 + kb;
    v8h lo = *(const v8h*)(p0);        // k = kb .. kb+7
    v8h hi = *(const v8h*)(p0 + 8);    // k = kb+8 .. kb+15
    v16h r;
#pragma unroll
    for (int i = 0; i < 8; ++i) { r[i] = lo[i]; r[8 + i] = hi[i]; }
    return r;
}

// ---------------------------------------------------------------------------
// Kernel 3: gather -> MLP (6->64->64->128, f16 WMMA, f32 accumulate) -> masked
// max-pool.  One workgroup per sample; each wave32 owns a 16-neighbor tile.
// Weights arrive as pre-swizzled fragments, staged once to LDS per block.
// ---------------------------------------------------------------------------
__global__ __launch_bounds__(256) void mlp_kernel(
    const float* __restrict__ x, const float* __restrict__ pos,
    const int* __restrict__ fps_idx, const int* __restrict__ nbr,
    const _Float16* __restrict__ wfrag,
    const float* __restrict__ b1, const float* __restrict__ b2,
    const float* __restrict__ b3,
    float* __restrict__ out)
{
    const int bs   = blockIdx.x;
    const int b    = bs >> 10;
    const int tid  = threadIdx.x;
    const int wave = tid >> 5;
    const int lane = tid & 31;

    __shared__ alignas(32) _Float16 wlds[NFRAG * 512];   // 28 KB fragments
    __shared__ alignas(16) _Float16 h[128 * 64];         // activation tiles
    __shared__ float    pmax[8 * 128];
    __shared__ int      vflag[128];
    __shared__ float    sq[3];
    __shared__ int      anyv;

    __builtin_amdgcn_s_wait_tensorcnt(0);

    if (tid == 0) {
        int qi = fps_idx[bs];
        const float* qp = pos + ((size_t)b * Nc + qi) * 3;
        sq[0] = qp[0]; sq[1] = qp[1]; sq[2] = qp[2];
        anyv = 0;
    }
    // cooperative 28KB weight-fragment stage (uint4 = 8 halfs per store)
    {
        const uint4* src = (const uint4*)wfrag;
        uint4* dst = (uint4*)wlds;
        for (int i = tid; i < NFRAG * 512 / 8; i += 256) dst[i] = src[i];
    }
    __syncthreads();

    // Layer-1 A fragment straight from the gather (6 feats, zero-pad K->32).
    // Lanes >=16 carry k=16..31 which are all zero for this layer.
    const int row = wave * 16 + (lane & 15);
    v16h a1 = zero_v16h();
    if (lane < 16) {
        int j = nbr[(size_t)bs * Kc + row];
        vflag[row] = (j >= 0) ? 1 : 0;
        if (j >= 0) {
            const float* xp = x   + ((size_t)b * Nc + j) * 3;
            const float* pp = pos + ((size_t)b * Nc + j) * 3;
            a1[0] = (_Float16)xp[0];
            a1[1] = (_Float16)xp[1];
            a1[2] = (_Float16)xp[2];
            a1[3] = (_Float16)(pp[0] - sq[0]);
            a1[4] = (_Float16)(pp[1] - sq[1]);
            a1[5] = (_Float16)(pp[2] - sq[2]);
        }
    }
    __syncthreads();
    if (tid < 128 && vflag[tid]) atomicOr(&anyv, 1);

    _Float16* hb = h + wave * 16 * 64;   // this wave's private tile

    // ---- layer 1: [16x6] x [6x64] + b1, ReLU ----
#pragma unroll
    for (int nt = 0; nt < 4; ++nt) {
        const float bias = b1[nt * 16 + (lane & 15)];
        v8f c;
#pragma unroll
        for (int jj = 0; jj < 8; ++jj) c[jj] = bias;
        c = __builtin_amdgcn_wmma_f32_16x16x32_f16(false, a1, false,
                wget(wlds, nt, lane), (short)0, c, false, false);
#pragma unroll
        for (int jj = 0; jj < 8; ++jj) {
            float v = fmaxf(c[jj], 0.0f);
            int m = jj + ((lane >> 4) << 3);
            hb[m * 64 + nt * 16 + (lane & 15)] = (_Float16)v;
        }
    }

    // ---- layer 2: [16x64] x [64x64] + b2, ReLU ----
    {
        v16h a0  = load_afrag(hb, 0,  lane);
        v16h a1k = load_afrag(hb, 32, lane);
        v8f c2[4];
#pragma unroll
        for (int nt = 0; nt < 4; ++nt) {
            const float bias = b2[nt * 16 + (lane & 15)];
            v8f c;
#pragma unroll
            for (int jj = 0; jj < 8; ++jj) c[jj] = bias;
            c = __builtin_amdgcn_wmma_f32_16x16x32_f16(false, a0, false,
                    wget(wlds, 4 + nt * 2 + 0, lane), (short)0, c, false, false);
            c = __builtin_amdgcn_wmma_f32_16x16x32_f16(false, a1k, false,
                    wget(wlds, 4 + nt * 2 + 1, lane), (short)0, c, false, false);
            c2[nt] = c;
        }
#pragma unroll
        for (int nt = 0; nt < 4; ++nt)
#pragma unroll
            for (int jj = 0; jj < 8; ++jj) {
                float v = fmaxf(c2[nt][jj], 0.0f);
                int m = jj + ((lane >> 4) << 3);
                hb[m * 64 + nt * 16 + (lane & 15)] = (_Float16)v;
            }
    }

    // ---- layer 3: [16x64] x [64x128] + b3, mask invalid rows, row-max ----
    {
        v16h a0  = load_afrag(hb, 0,  lane);
        v16h a1k = load_afrag(hb, 32, lane);
#pragma unroll
        for (int nt = 0; nt < 8; ++nt) {
            const float bias = b3[nt * 16 + (lane & 15)];
            v8f c;
#pragma unroll
            for (int jj = 0; jj < 8; ++jj) c[jj] = bias;
            c = __builtin_amdgcn_wmma_f32_16x16x32_f16(false, a0, false,
                    wget(wlds, 12 + nt * 2 + 0, lane), (short)0, c, false, false);
            c = __builtin_amdgcn_wmma_f32_16x16x32_f16(false, a1k, false,
                    wget(wlds, 12 + nt * 2 + 1, lane), (short)0, c, false, false);
            float m8 = -INFINITY;
#pragma unroll
            for (int jj = 0; jj < 8; ++jj) {
                int r = wave * 16 + jj + ((lane >> 4) << 3);
                float v = vflag[r] ? c[jj] : -INFINITY;
                m8 = fmaxf(m8, v);
            }
            float o = fmaxf(m8, __shfl_xor(m8, 16));   // combine row halves
            if (lane < 16) pmax[wave * 128 + nt * 16 + lane] = o;
        }
    }
    __syncthreads();
    if (tid < 128) {
        float v = pmax[tid];
#pragma unroll
        for (int w = 1; w < 8; ++w) v = fmaxf(v, pmax[w * 128 + tid]);
        out[(size_t)bs * 128 + tid] = anyv ? v : 0.0f;
    }
}

// ---------------------------------------------------------------------------
// Kernel 4: pos_out / batch_out pass-throughs
// ---------------------------------------------------------------------------
__global__ __launch_bounds__(256) void tail_kernel(const float* __restrict__ pos,
                                                   const int* __restrict__ batch,
                                                   const int* __restrict__ fps_idx,
                                                   float* __restrict__ out_pos,
                                                   float* __restrict__ out_batch)
{
    int t = blockIdx.x * 256 + threadIdx.x;
    if (t >= Bc * Sc) return;
    int b  = t >> 10;
    int qi = fps_idx[t];
    const float* pp = pos + ((size_t)b * Nc + qi) * 3;
    out_pos[3*t+0] = pp[0];
    out_pos[3*t+1] = pp[1];
    out_pos[3*t+2] = pp[2];
    out_batch[t] = (float)batch[(size_t)b * Nc + qi];
}

// ---------------------------------------------------------------------------
extern "C" void kernel_launch(void* const* d_in, const int* in_sizes, int n_in,
                              void* d_out, int out_size, void* d_ws, size_t ws_size,
                              hipStream_t stream)
{
    const float* x     = (const float*)d_in[0];
    const float* pos   = (const float*)d_in[1];
    const float* W1    = (const float*)d_in[2];
    const float* b1    = (const float*)d_in[3];
    const float* W2    = (const float*)d_in[4];
    const float* b2    = (const float*)d_in[5];
    const float* W3    = (const float*)d_in[6];
    const float* b3    = (const float*)d_in[7];
    const int*   batch = (const int*)d_in[8];
    float* out = (float*)d_out;

    // workspace layout: fps indices | neighbor lists | swizzled weight frags
    char* ws = (char*)d_ws;
    int* fps_idx = (int*)ws;
    size_t off1 = ((size_t)Bc * Sc * sizeof(int) + 255) & ~(size_t)255;
    int* nbr = (int*)(ws + off1);
    size_t off2 = off1 + (((size_t)Bc * Sc * Kc * sizeof(int) + 255) & ~(size_t)255);
    _Float16* wfrag = (_Float16*)(ws + off2);

    prep_kernel<<<(NFRAG * 512 + 255) / 256, 256, 0, stream>>>(W1, W2, W3, wfrag);
    fps_kernel <<<Bc,       256, 0, stream>>>(pos, fps_idx);
    ball_kernel<<<Bc * Sc,  256, 0, stream>>>(pos, fps_idx, nbr);
    mlp_kernel <<<Bc * Sc,  256, 0, stream>>>(x, pos, fps_idx, nbr, wfrag,
                                              b1, b2, b3, out);
    tail_kernel<<<(Bc*Sc + 255)/256, 256, 0, stream>>>(
        pos, batch, fps_idx,
        out + (size_t)Bc * Sc * 128,
        out + (size_t)Bc * Sc * 131);
}
